// SimpleFiBiNET_82471962018426
// MI455X (gfx1250) — compile-verified, matrix operand order checked by reading, not run on hardware
//
#include <hip/hip_runtime.h>
#include <hip/hip_bf16.h>

// ---------------------------------------------------------------------------
// SimpleFiBiNET forward for MI455X (gfx1250), bf16 WMMA pipeline.
//
// Shapes: B=4096, F=32, D=64, P=F*(F-1)/2=496, NIN=F*D+P*D=33792,
//         MLP 33792->256->128->64->1, RED=10, VOCAB=10000.
//
// Algebraic restructure of the bilinear interaction:
//   inter[b,p,o] = sum_{d,k} V[b,i,d] * bil_w[o,d,k] * V[b,j,k]
//   => U[b,j,(o,d)] = sum_k bil_w[(o,d),k] * V[b,j,k]        (68.7 GFLOP)
//      inter[b,p,o] = sum_d V[b,i,d] * U[b,j,o,d]            (16.6 GFLOP)
// instead of the naive 1.07 TFLOP. U lives only in LDS: one workgroup per
// batch row keeps V (4KB) + the FULL 32-row U slab (256KB bf16) resident
// (CDNA5: 320KB LDS/WGP) so bil_w is streamed from L2 exactly once per row.
//
// All heavy math uses v_wmma_f32_16x16x32_bf16. Fragment element layout per
// CDNA5 ISA (16-bit A 16x32): lane L holds row m=L%16, elements e=0..15 map
// to k = (e<8?0:16) + (L/16)*8 + e%8 -> two contiguous runs of 8 bf16 per
// lane -> every fragment load is 2x 16-byte (b128) loads. B assumed mirrored
// (lane col n=L%16, same k runs). C/D f32: lane col n=L%16, row m=(L/16)*8+r.
//
// MLP GEMMs use a 2x4 register block per wave (32x64 output tile): per
// k-step 6 fragment loads feed 8 WMMAs (~21 FLOP/B from L2) with fused
// bias+BN+ReLU epilogue.
// ---------------------------------------------------------------------------

#define B_SZ   4096
#define F_SZ   32
#define D_SZ   64
#define VOCAB  10000
#define RED    10
#define NIN    33792      // F*D + P*D
#define VOFF   2048       // F*D, start of inter region within an X row
#define H1     256
#define H2     128
#define H3     64
#define BN_EPS 1e-5f

typedef __attribute__((ext_vector_type(16))) __bf16 v16bf;
typedef __attribute__((ext_vector_type(8)))  __bf16 bf16x8;
typedef __attribute__((ext_vector_type(8)))  float  v8f;

// Load one wave-lane's 16 bf16 fragment elements: two contiguous runs of 8
// (runs are 16 k-apart). base0 already includes kbase + (lane/16)*8.
__device__ __forceinline__ v16bf load_frag(const __bf16* base0) {
  bf16x8 lo = *(const bf16x8*)(base0);
  bf16x8 hi = *(const bf16x8*)(base0 + 16);
  v16bf r;
#pragma unroll
  for (int e = 0; e < 8; ++e) { r[e] = lo[e]; r[e + 8] = hi[e]; }
  return r;
}

__device__ __forceinline__ v8f wmma_bf16(v16bf a, v16bf b, v8f c) {
  // (neg_a, A, neg_b, B, c_mod, C, reuse_a, reuse_b)
  return __builtin_amdgcn_wmma_f32_16x16x32_bf16(false, a, false, b,
                                                 (short)0, c, false, false);
}

// ---------------------------------------------------------------------------
// K0: f32 -> bf16 grid-stride convert (weights).
// ---------------------------------------------------------------------------
__global__ void f32_to_bf16_kernel(const float* __restrict__ in,
                                   __bf16* __restrict__ out, int n) {
  int i = blockIdx.x * blockDim.x + threadIdx.x;
  int stride = gridDim.x * blockDim.x;
  for (; i < n; i += stride) out[i] = (__bf16)in[i];
}

// ---------------------------------------------------------------------------
// K1: embedding gather + SENet + reweight. One workgroup (256 thr) per batch
// row. Writes V (bf16) into cols [0,2048) of X row b. Gather is served from
// L2 (table = 82MB < 192MB L2).
// ---------------------------------------------------------------------------
__global__ void embed_senet_kernel(const int*   __restrict__ cat,
                                   const float* __restrict__ emb,
                                   const float* __restrict__ se_w1,
                                   const float* __restrict__ se_b1,
                                   const float* __restrict__ se_w2,
                                   const float* __restrict__ se_b2,
                                   __bf16* __restrict__ Xbf) {
  __shared__ float Vr[F_SZ][D_SZ];
  __shared__ float Z[F_SZ];
  __shared__ float Hr[16];
  __shared__ float Af[F_SZ];
  const int b = blockIdx.x;
  const int t = threadIdx.x;
  const int fo = t >> 6;        // 0..3
  const int d  = t & 63;

#pragma unroll
  for (int g = 0; g < 8; ++g) {
    const int f   = g * 4 + fo;
    const int idx = cat[b * F_SZ + f];
    Vr[f][d] = emb[((size_t)f * VOCAB + (size_t)idx) * D_SZ + d];
  }
  __syncthreads();
  if (t < F_SZ) {               // Z = mean over D
    float s = 0.f;
#pragma unroll
    for (int k = 0; k < D_SZ; ++k) s += Vr[t][k];
    Z[t] = s * (1.f / 64.f);
  }
  __syncthreads();
  if (t < RED) {                // relu(Z @ w1^T + b1)
    float s = se_b1[t];
#pragma unroll
    for (int f = 0; f < F_SZ; ++f) s += Z[f] * se_w1[t * F_SZ + f];
    Hr[t] = fmaxf(s, 0.f);
  }
  __syncthreads();
  if (t < F_SZ) {               // sigmoid(h @ w2^T + b2)
    float s = se_b2[t];
#pragma unroll
    for (int r = 0; r < RED; ++r) s += Hr[r] * se_w2[t * RED + r];
    Af[t] = 1.f / (1.f + __expf(-s));
  }
  __syncthreads();
  __bf16* Xrow = Xbf + (size_t)b * NIN;
#pragma unroll
  for (int g = 0; g < 8; ++g) {
    const int f = g * 4 + fo;
    Xrow[f * D_SZ + d] = (__bf16)(Vr[f][d] * Af[f]);
  }
}

// ---------------------------------------------------------------------------
// K2: fused bilinear interaction. One workgroup (8 waves) per batch row b.
//   Stage 1: U[j][od] = sum_k V[j][k] * bil_w[od][k]
//            WMMA M=32 (2 m-tiles, A frags reused across 32 n-tiles/wave),
//            each bil_w B-fragment loaded ONCE and used by both m-tiles.
//   Stage 2: inter[p(i,j),o] = sum_d V[i][d] * U[j][o*64+d] (+ bil_b)
// dyn LDS = 32*64*2 (V) + 32*4096*2 (U) = 266240 B  (<= 320KB/WGP).
// ---------------------------------------------------------------------------
__global__ void bilinear_kernel(__bf16* __restrict__ Xbf,
                                const __bf16* __restrict__ Wbil,   // [od=4096][k=64]
                                const float* __restrict__ bil_b) {
  extern __shared__ __bf16 smem[];
  __bf16* Vl = smem;                    // [32][64]
  __bf16* Ul = smem + F_SZ * D_SZ;      // [32][4096]
  const int b    = blockIdx.x;
  const int t    = threadIdx.x;
  const int lane = t & 31;
  const int wave = t >> 5;
  const int half = lane >> 4;           // 0/1
  const int lm   = lane & 15;
  __bf16* Xrow = Xbf + (size_t)b * NIN;

  // stage in V[b] (2048 bf16): one b128 per thread
  *(bf16x8*)(Vl + t * 8) = *(const bf16x8*)(Xrow + t * 8);
  __syncthreads();

  // ---- Stage 1: A frags for both m-tiles & both k-steps (reused 32x) ----
  const v16bf a00 = load_frag(Vl + lm * D_SZ + 0 + half * 8);         // j 0..15
  const v16bf a01 = load_frag(Vl + lm * D_SZ + 32 + half * 8);
  const v16bf a10 = load_frag(Vl + (16 + lm) * D_SZ + 0 + half * 8);  // j 16..31
  const v16bf a11 = load_frag(Vl + (16 + lm) * D_SZ + 32 + half * 8);
  for (int it = 0; it < 32; ++it) {          // 256 n-tiles over 8 waves
    const int n = (wave * 32 + it) * 16 + lm;   // od column
    v8f c0 = {}, c1 = {};
    const v16bf b0 = load_frag(Wbil + (size_t)n * D_SZ + 0 + half * 8);
    c0 = wmma_bf16(a00, b0, c0);
    c1 = wmma_bf16(a10, b0, c1);
    const v16bf b1 = load_frag(Wbil + (size_t)n * D_SZ + 32 + half * 8);
    c0 = wmma_bf16(a01, b1, c0);
    c1 = wmma_bf16(a11, b1, c1);
#pragma unroll
    for (int r = 0; r < 8; ++r) {
      Ul[(half * 8 + r) * 4096 + n]        = (__bf16)c0[r];
      Ul[(16 + half * 8 + r) * 4096 + n]   = (__bf16)c1[r];
    }
  }
  __syncthreads();

  // ---- Stage 2: j's strided over waves; GEMM (i x o) = V[0..j) x U[j]^T ----
  for (int j = wave + 1; j < F_SZ; j += 8) {
    const int mtiles = (j + 15) >> 4;
    for (int mt = 0; mt < mtiles; ++mt) {
      const v16bf a20 = load_frag(Vl + (mt * 16 + lm) * D_SZ + 0 + half * 8);
      const v16bf a21 = load_frag(Vl + (mt * 16 + lm) * D_SZ + 32 + half * 8);
#pragma unroll
      for (int nt = 0; nt < 4; ++nt) {
        const int o = nt * 16 + lm;
        v8f c = {};
        c = wmma_bf16(a20, load_frag(Ul + (size_t)j * 4096 + o * D_SZ + 0 + half * 8), c);
        c = wmma_bf16(a21, load_frag(Ul + (size_t)j * 4096 + o * D_SZ + 32 + half * 8), c);
        const float bb = bil_b[o];
#pragma unroll
        for (int r = 0; r < 8; ++r) {
          const int i = mt * 16 + half * 8 + r;
          if (i < j) {
            const int p = 31 * i - (i * (i - 1)) / 2 + (j - i - 1);
            Xrow[VOFF + p * D_SZ + o] = (__bf16)(c[r] + bb);
          }
        }
      }
    }
  }
}

// ---------------------------------------------------------------------------
// K3: GEMM (out = relu(BN(A @ B^T + bias))) in bf16 WMMA, f32 accumulate.
// A: [M][K] bf16 (row stride K). Bw: [N][K] bf16 (W row-major => B[k][n]
// contiguous along k for fixed n).
// Register blocking: each wave owns a 32x64 output tile (2 m-tiles x 4
// n-tiles): per k-step 6 fragment loads drive 8 WMMAs. 8 waves/block.
// Requires: M % 32 == 0, N % 64 == 0, K % 32 == 0.
// ---------------------------------------------------------------------------
__global__ void gemm_bn_relu_kernel(const __bf16* __restrict__ A,
                                    const __bf16* __restrict__ Bw,
                                    const float* __restrict__ bias,
                                    const float* __restrict__ gamma,
                                    const float* __restrict__ beta,
                                    const float* __restrict__ rmean,
                                    const float* __restrict__ rvar,
                                    __bf16* __restrict__ out,
                                    int M, int N, int K) {
  const int t    = threadIdx.x;
  const int lane = t & 31;
  const int wave = t >> 5;
  const int half = lane >> 4;
  const int lm   = lane & 15;
  const int col_groups = N >> 6;              // 64-wide column groups
  const int wt = blockIdx.x * 8 + wave;
  const int cg = wt % col_groups;
  const int rg = wt / col_groups;             // 32-row groups

  const size_t  row16 = (size_t)16 * K;
  const __bf16* Abase = A  + (size_t)(rg * 32 + lm) * K + half * 8;
  const __bf16* Bbase = Bw + (size_t)(cg * 64 + lm) * K + half * 8;

  v8f c[2][4] = {};
  for (int kb = 0; kb < K; kb += 32) {
    if (kb + 1024 < K) {   // stream hints -> global_prefetch_b8
      __builtin_prefetch(Abase + kb + 1024, 0, 1);
      __builtin_prefetch(Bbase + kb + 1024, 0, 1);
    }
    const v16bf a0 = load_frag(Abase + kb);
    const v16bf a1 = load_frag(Abase + row16 + kb);
#pragma unroll
    for (int nt = 0; nt < 4; ++nt) {
      const v16bf b = load_frag(Bbase + (size_t)nt * row16 + kb);
      c[0][nt] = wmma_bf16(a0, b, c[0][nt]);
      c[1][nt] = wmma_bf16(a1, b, c[1][nt]);
    }
  }

#pragma unroll
  for (int nt = 0; nt < 4; ++nt) {
    const int n = cg * 64 + nt * 16 + lm;
    const float sc = rsqrtf(rvar[n] + BN_EPS) * gamma[n];
    const float sh = beta[n] - rmean[n] * sc + bias[n] * sc;
#pragma unroll
    for (int mt = 0; mt < 2; ++mt) {
#pragma unroll
      for (int r = 0; r < 8; ++r) {
        const int m = rg * 32 + mt * 16 + half * 8 + r;
        float h = c[mt][nt][r] * sc + sh;     // ((acc+bias)-rm)*rsqrt*g+beta
        h = fmaxf(h, 0.f);
        out[(size_t)m * N + n] = (__bf16)h;
      }
    }
  }
}

// ---------------------------------------------------------------------------
// K4: final 64-wide dot per row.
// ---------------------------------------------------------------------------
__global__ void final_kernel(const __bf16* __restrict__ X3,
                             const float* __restrict__ Wf,
                             const float* __restrict__ bf_,
                             float* __restrict__ out) {
  const int b = blockIdx.x * blockDim.x + threadIdx.x;
  if (b < B_SZ) {
    float s = bf_[0];
    const __bf16* xr = X3 + (size_t)b * H3;
#pragma unroll
    for (int k = 0; k < H3; ++k) s += (float)xr[k] * Wf[k];
    out[b] = s;
  }
}

// ---------------------------------------------------------------------------
// host launcher
// ---------------------------------------------------------------------------
static inline size_t ws_alloc(size_t& off, size_t bytes) {
  size_t o = off;
  off = (off + bytes + 255) & ~(size_t)255;
  return o;
}

extern "C" void kernel_launch(void* const* d_in, const int* in_sizes, int n_in,
                              void* d_out, int out_size, void* d_ws, size_t ws_size,
                              hipStream_t stream) {
  (void)in_sizes; (void)n_in; (void)out_size; (void)ws_size;
  const int*   cat    = (const int*)  d_in[0];
  // d_in[1] = numerical_x : unused by the reference
  const float* emb    = (const float*)d_in[2];
  const float* se_w1  = (const float*)d_in[3];
  const float* se_b1  = (const float*)d_in[4];
  const float* se_w2  = (const float*)d_in[5];
  const float* se_b2  = (const float*)d_in[6];
  const float* bil_w  = (const float*)d_in[7];   // [o][d][k] == [od][k]
  const float* bil_b  = (const float*)d_in[8];
  const float* W1 = (const float*)d_in[9],  *b1 = (const float*)d_in[10];
  const float* g1 = (const float*)d_in[11], *be1 = (const float*)d_in[12];
  const float* rm1 = (const float*)d_in[13], *rv1 = (const float*)d_in[14];
  const float* W2 = (const float*)d_in[15], *b2 = (const float*)d_in[16];
  const float* g2 = (const float*)d_in[17], *be2 = (const float*)d_in[18];
  const float* rm2 = (const float*)d_in[19], *rv2 = (const float*)d_in[20];
  const float* W3 = (const float*)d_in[21], *b3 = (const float*)d_in[22];
  const float* g3 = (const float*)d_in[23], *be3 = (const float*)d_in[24];
  const float* rm3 = (const float*)d_in[25], *rv3 = (const float*)d_in[26];
  const float* Wf = (const float*)d_in[27], *bf_ = (const float*)d_in[28];
  float* out = (float*)d_out;

  // workspace layout (~298 MB total, bf16 activations/weights)
  char* ws = (char*)d_ws;
  size_t off = 0;
  __bf16* Xbf   = (__bf16*)(ws + ws_alloc(off, (size_t)B_SZ * NIN * 2));
  __bf16* W1bf  = (__bf16*)(ws + ws_alloc(off, (size_t)H1 * NIN * 2));
  __bf16* W2bf  = (__bf16*)(ws + ws_alloc(off, (size_t)H2 * H1 * 2));
  __bf16* W3bf  = (__bf16*)(ws + ws_alloc(off, (size_t)H3 * H2 * 2));
  __bf16* Wbil  = (__bf16*)(ws + ws_alloc(off, (size_t)D_SZ * D_SZ * D_SZ * 2));
  __bf16* X1    = (__bf16*)(ws + ws_alloc(off, (size_t)B_SZ * H1 * 2));
  __bf16* X2    = (__bf16*)(ws + ws_alloc(off, (size_t)B_SZ * H2 * 2));
  __bf16* X3    = (__bf16*)(ws + ws_alloc(off, (size_t)B_SZ * H3 * 2));

  // K0: weight converts (cheap; deterministic each call)
  {
    const int nW1 = H1 * NIN;
    f32_to_bf16_kernel<<<2048, 256, 0, stream>>>(W1, W1bf, nW1);
    f32_to_bf16_kernel<<<128, 256, 0, stream>>>(W2, W2bf, H2 * H1);
    f32_to_bf16_kernel<<<32, 256, 0, stream>>>(W3, W3bf, H3 * H2);
    f32_to_bf16_kernel<<<1024, 256, 0, stream>>>(bil_w, Wbil, D_SZ * D_SZ * D_SZ);
  }

  // K1: gather + SENet + scale -> V region of X
  embed_senet_kernel<<<B_SZ, 256, 0, stream>>>(cat, emb, se_w1, se_b1,
                                               se_w2, se_b2, Xbf);

  // K2: fused bilinear -> inter region of X (266240 B dynamic LDS)
  bilinear_kernel<<<B_SZ, 256, 266240, stream>>>(Xbf, Wbil, bil_b);

  // K3: MLP (each wave: 32x64 tile; 8 waves/block)
  gemm_bn_relu_kernel<<<(B_SZ / 32) * (H1 / 64) / 8, 256, 0, stream>>>(
      Xbf, W1bf, b1, g1, be1, rm1, rv1, X1, B_SZ, H1, NIN);
  gemm_bn_relu_kernel<<<(B_SZ / 32) * (H2 / 64) / 8, 256, 0, stream>>>(
      X1, W2bf, b2, g2, be2, rm2, rv2, X2, B_SZ, H2, H1);
  gemm_bn_relu_kernel<<<(B_SZ / 32) * (H3 / 64) / 8, 256, 0, stream>>>(
      X2, W3bf, b3, g3, be3, rm3, rv3, X3, B_SZ, H3, H2);

  // K4: final projection
  final_kernel<<<B_SZ / 256, 256, 0, stream>>>(X3, Wf, bf_, out);
}
